// LSTM_Decoder_19078244729397
// MI455X (gfx1250) — matrix operand order repaired and checked by reading, not compiled, not run on hardware
//
#include <hip/hip_runtime.h>
#include <math.h>

// Problem constants (from reference): B=32, S=1024, D=H=512, L=2, gates=4H=2048.
#define BATCH 32
#define SEQ   1024
#define HID   512
#define GATES 2048
#define NWG   32      // persistent workgroups; each owns 16 hidden columns (=64 gate cols)
#define NTHR  256     // 8 waves of 32

typedef __attribute__((ext_vector_type(16))) __bf16          v16bf;
typedef __attribute__((ext_vector_type(16))) unsigned short  v16us;
typedef __attribute__((ext_vector_type(8)))  unsigned short  v8us;
typedef __attribute__((ext_vector_type(8)))  float           v8f;

// ---------- f32 -> bf16 (round to nearest even) -- cold paths only ----------
__device__ __forceinline__ unsigned short f2bf(float f) {
  unsigned int u = __float_as_uint(f);
  u += 0x7FFFu + ((u >> 16) & 1u);
  return (unsigned short)(u >> 16);
}

// ---------- A-fragment loader: 16x32 bf16, contiguous k order (as packed B), pure b128 loads ----
// lane L (<16): M=L,    K = {0..15} + chunk base
// lane L (>=16): M=L-16, K = {16..31} + chunk base
__device__ __forceinline__ v16bf load_a(const unsigned short* __restrict__ base, long stride,
                                        int k0, int mtile, int lane) {
  int m = (mtile << 4) + (lane & 15);
  const unsigned short* r = base + (long)m * stride + k0 + ((lane & 16) ? 16 : 0);
  v8us lo = *(const v8us*)(r);        // 16B
  v8us hi = *(const v8us*)(r + 8);    // 16B
  v16us q = __builtin_shufflevector(lo, hi, 0, 1, 2, 3, 4, 5, 6, 7,
                                    8, 9, 10, 11, 12, 13, 14, 15);
  return __builtin_bit_cast(v16bf, q);
}

// ---------- B-fragment loader from LDS (pre-swizzled at pack time) ----------
// block `blk` = gate*KS + ks; each lane reads its own contiguous 16 bf16 (32B).
__device__ __forceinline__ v16bf load_b(const unsigned short* w, int blk, int lane) {
  const v16us* p = (const v16us*)(w + (((blk << 5) + lane) << 4));
  return __builtin_bit_cast(v16bf, *p);
}

__device__ __forceinline__ v8f mm_accum(v8f acc, const unsigned short* __restrict__ abase,
                                        long astride, const unsigned short* ldsw, int gate,
                                        int KS, int ksbeg, int ksend, int mtile, int lane) {
#pragma unroll 8
  for (int ks = ksbeg; ks < ksend; ++ks) {
    v16bf a = load_a(abase, astride, (ks - ksbeg) * 32, mtile, lane);
    v16bf b = load_b(ldsw, gate * KS + ks, lane);
    acc = __builtin_amdgcn_wmma_f32_16x16x32_bf16(false, a, false, b, (short)0, acc,
                                                  false, false);
  }
  return acc;
}

__device__ __forceinline__ float sigf(float x) { return 1.0f / (1.0f + __expf(-x)); }

// LSTM cell elementwise for one (batch row b, local hidden col hc); c updated in place.
__device__ __forceinline__ float cellf(const float* gb, const float* bias, int b, int hc,
                                       float& c) {
  const float* g = gb + b * 64 + hc;
  float gi = g[0]  + bias[hc];
  float gf = g[16] + bias[16 + hc];
  float gg = g[32] + bias[32 + hc];
  float go = g[48] + bias[48 + hc];
  c = sigf(gf) * c + sigf(gi) * tanhf(gg);
  return sigf(go) * tanhf(c);
}

// Grid-wide barrier: monotonic counter, release add + acquire spin.
__device__ __forceinline__ void grid_barrier(unsigned int* ctr, unsigned int target) {
  __syncthreads();
  __threadfence();
  if (threadIdx.x == 0) {
    __hip_atomic_fetch_add(ctr, 1u, __ATOMIC_RELEASE, __HIP_MEMORY_SCOPE_AGENT);
    while (__hip_atomic_load(ctr, __ATOMIC_ACQUIRE, __HIP_MEMORY_SCOPE_AGENT) < target) {
      __builtin_amdgcn_s_sleep(1);
    }
  }
  __syncthreads();
}

// =====================================================================
// Kernel 0: convert x [B,S,D] f32 -> bf16 (read once per step afterwards)
// =====================================================================
__global__ void x_prep(const float* __restrict__ x, unsigned short* __restrict__ xp) {
  long i = ((long)blockIdx.x * NTHR + threadIdx.x) * 8;   // 16777216 elems total
  float4 a = *(const float4*)(x + i);
  float4 b = *(const float4*)(x + i + 4);
  v8us q;
  q[0] = f2bf(a.x); q[1] = f2bf(a.y); q[2] = f2bf(a.z); q[3] = f2bf(a.w);
  q[4] = f2bf(b.x); q[5] = f2bf(b.y); q[6] = f2bf(b.z); q[7] = f2bf(b.w);
  *(v8us*)(xp + i) = q;
}

// =====================================================================
// Kernel 1: pack W into per-WG, per-(gate,kstep), per-lane bf16 blocks
// in the contiguous k order load_a uses: lane L -> N=L%16, K=16*(L/16)+e.
// Layout per WG (131072 bf16): [W_ih0 32768][W_hh0 32768][W1 (ih1|hh1) 65536]
// =====================================================================
__global__ void lstm_prep(const float* __restrict__ W_ih, const float* __restrict__ W_hh,
                          unsigned short* __restrict__ wpack) {
  int t = blockIdx.x * NTHR + threadIdx.x;   // 262144 slots, 16 bf16 each
  int wg = t >> 13;                          // 8192 slots per WG
  int s  = t & 8191;
  int sec, sl;
  if (s < 2048)      { sec = 0; sl = s; }
  else if (s < 4096) { sec = 1; sl = s - 2048; }
  else               { sec = 2; sl = s - 4096; }
  int lane = sl & 31;
  int blk  = sl >> 5;
  int KS   = (sec == 2) ? 32 : 16;
  int g    = blk / KS;
  int ks   = blk % KS;
  int grow  = g * HID + wg * 16 + (lane & 15);   // gate row in [4H]
  int kbase = ks * 32 + (lane & 16);             // K chunk, 16 contiguous elems
  const float* src;
  if (sec == 0)      src = W_ih + (long)grow * 512 + kbase;
  else if (sec == 1) src = W_hh + (long)grow * 512 + kbase;
  else {
    if (kbase < 512) src = W_ih + 2048L * 512 + (long)grow * 512 + kbase;
    else             src = W_hh + 2048L * 512 + (long)grow * 512 + (kbase - 512);
  }
  unsigned short* dst =
      wpack + (long)wg * 131072 + (long)(sec == 0 ? 0 : (sec == 1 ? 32768 : 65536)) +
      (long)sl * 16;
#pragma unroll
  for (int e = 0; e < 16; ++e) dst[e] = f2bf(src[e]);
}

// =====================================================================
// Kernel 2: init combined biases, h state (both layers = enc_h, bf16), counter
// =====================================================================
__global__ void lstm_init(const float* __restrict__ enc_h, const float* __restrict__ b_ih,
                          const float* __restrict__ b_hh, unsigned short* __restrict__ act,
                          float* __restrict__ biaspack, unsigned int* __restrict__ ctr) {
  int tid = blockIdx.x * NTHR + threadIdx.x;   // 32768 threads
  if (tid == 0) *ctr = 0u;
  if (tid < 4096) {
    int wg = tid >> 7, r = tid & 127, layer = r >> 6, gc = r & 63;
    int row = (gc >> 4) * HID + wg * 16 + (gc & 15);
    biaspack[tid] = b_ih[layer * GATES + row] + b_hh[layer * GATES + row];
  }
  // act buffer 0, layers 0 & 1: broadcast enc_h (bf16)
  act[tid] = f2bf(enc_h[tid & 16383]);
}

// =====================================================================
// Kernel 3: persistent LSTM. 32 WGs x 256 threads, weights in LDS.
// LDS: [W0x 64KB][W0h 64KB][W1 128KB][gatebuf 8KB][bias 512B] = 270848B
// =====================================================================
__global__ __launch_bounds__(NTHR, 1) void lstm_persist(
    const unsigned short* __restrict__ xp, const float* __restrict__ enc_c,
    const unsigned short* __restrict__ wpack, const float* __restrict__ biaspack,
    unsigned short* __restrict__ act, unsigned int* __restrict__ ctr,
    float* __restrict__ out) {
  extern __shared__ char smem[];
  unsigned short* ldsW    = (unsigned short*)smem;       // W0x @0, W0h @32768, W1 @65536
  float*          gatebuf = (float*)(smem + 262144);     // [32][4*16] f32
  float*          ldsbias = (float*)(smem + 262144 + 8192);  // [2][64]

  const int tid  = threadIdx.x;
  const int lane = tid & 31;
  const int wave = tid >> 5;
  const int mtile = wave & 1;    // batch rows 0-15 / 16-31
  const int gate  = wave >> 1;   // 0=i 1=f 2=g 3=o
  const int wg    = blockIdx.x;

  // ---- stage this WG's weight slice into LDS (256KB, b128 copies) ----
  {
    const uint4* src = (const uint4*)(wpack + (size_t)wg * 131072);
    uint4* dst = (uint4*)smem;
    for (int i = tid; i < 16384; i += NTHR) dst[i] = src[i];
    if (tid < 128) ldsbias[tid] = biaspack[wg * 128 + tid];
  }
  __syncthreads();

  // ---- per-thread cell state: (b0,hc) and (b0+16,hc) of this WG's 16 cols ----
  const int b0  = tid >> 4;        // 0..15
  const int hc  = tid & 15;
  const int col = wg * 16 + hc;    // global hidden column
  float c0_a = enc_c[b0 * HID + col];
  float c0_b = enc_c[(b0 + 16) * HID + col];
  float c1_a = c0_a, c1_b = c0_b;

  float* hfin = out + (long)BATCH * SEQ * HID;
  float* cfin = hfin + BATCH * HID;

  const int am = (mtile << 4) + (lane & 15);   // this lane's batch row (for prefetch)

#pragma unroll 1
  for (int t = 0; t < SEQ; ++t) {
    const int p = t & 1, np = p ^ 1;
    const unsigned short* h0prev = act + (p * 2 + 0) * 16384;
    const unsigned short* h1prev = act + (p * 2 + 1) * 16384;
    unsigned short*       h0cur  = act + (np * 2 + 0) * 16384;
    unsigned short*       h1cur  = act + (np * 2 + 1) * 16384;

    // prefetch next step's x row for this lane (global_prefetch path)
    if (t + 1 < SEQ) {
      __builtin_prefetch(xp + (long)am * (SEQ * 512) + (long)(t + 1) * 512, 0, 0);
    }

    // ================= layer 0: gates = x_t@Wih0^T + h0@Whh0^T =================
    v8f acc = {};
    acc = mm_accum(acc, xp + (long)t * 512, (long)SEQ * 512, ldsW, gate, 16, 0, 16,
                   mtile, lane);
    acc = mm_accum(acc, h0prev, 512L, ldsW + 32768, gate, 16, 0, 16, mtile, lane);
#pragma unroll
    for (int r = 0; r < 8; ++r) {
      int brow = mtile * 16 + r + ((lane & 16) ? 8 : 0);
      gatebuf[brow * 64 + gate * 16 + (lane & 15)] = acc[r];
    }
    __syncthreads();
    {
      float h = cellf(gatebuf, ldsbias, b0, hc, c0_a);
      h0cur[b0 * HID + col] = f2bf(h);
      h = cellf(gatebuf, ldsbias, b0 + 16, hc, c0_b);
      h0cur[(b0 + 16) * HID + col] = f2bf(h);
    }
    grid_barrier(ctr, (unsigned)(NWG * (2 * t + 1)));

    // ================= layer 1: gates = h0_cur@Wih1^T + h1_prev@Whh1^T =========
    acc = (v8f){};
    acc = mm_accum(acc, h0cur,  512L, ldsW + 65536, gate, 32, 0, 16, mtile, lane);
    acc = mm_accum(acc, h1prev, 512L, ldsW + 65536, gate, 32, 16, 32, mtile, lane);
#pragma unroll
    for (int r = 0; r < 8; ++r) {
      int brow = mtile * 16 + r + ((lane & 16) ? 8 : 0);
      gatebuf[brow * 64 + gate * 16 + (lane & 15)] = acc[r];
    }
    __syncthreads();
    {
      float h = cellf(gatebuf, ldsbias + 64, b0, hc, c1_a);
      h1cur[b0 * HID + col] = f2bf(h);
      out[(long)b0 * SEQ * HID + (long)t * HID + col] = h;
      if (t == SEQ - 1) hfin[b0 * HID + col] = h;
      h = cellf(gatebuf, ldsbias + 64, b0 + 16, hc, c1_b);
      h1cur[(b0 + 16) * HID + col] = f2bf(h);
      out[(long)(b0 + 16) * SEQ * HID + (long)t * HID + col] = h;
      if (t == SEQ - 1) hfin[(b0 + 16) * HID + col] = h;
    }
    grid_barrier(ctr, (unsigned)(NWG * (2 * t + 2)));
  }

  cfin[b0 * HID + col] = c1_a;
  cfin[(b0 + 16) * HID + col] = c1_b;
}

// =====================================================================
extern "C" void kernel_launch(void* const* d_in, const int* in_sizes, int n_in,
                              void* d_out, int out_size, void* d_ws, size_t ws_size,
                              hipStream_t stream) {
  const float* x     = (const float*)d_in[0];
  const float* enc_h = (const float*)d_in[3];
  const float* enc_c = (const float*)d_in[4];
  const float* W_ih  = (const float*)d_in[5];
  const float* W_hh  = (const float*)d_in[6];
  const float* b_ih  = (const float*)d_in[7];
  const float* b_hh  = (const float*)d_in[8];

  // workspace: [xpack 33554432B][wpack 8388608B][biaspack 16384B][act bf16 131072B][ctr]
  unsigned short* xpack    = (unsigned short*)d_ws;
  unsigned short* wpack    = (unsigned short*)((char*)d_ws + 33554432);
  float*          biaspack = (float*)((char*)d_ws + 33554432 + 8388608);
  unsigned short* act      = (unsigned short*)((char*)d_ws + 33554432 + 8388608 + 16384);
  unsigned int*   ctr      = (unsigned int*)((char*)d_ws + 33554432 + 8388608 + 16384 + 131072);

  float* out = (float*)d_out;

  x_prep<<<8192, NTHR, 0, stream>>>(x, xpack);
  lstm_prep<<<1024, NTHR, 0, stream>>>(W_ih, W_hh, wpack);
  lstm_init<<<128, NTHR, 0, stream>>>(enc_h, b_ih, b_hh, act, biaspack, ctr);
  lstm_persist<<<NWG, NTHR, 270848, stream>>>(xpack, enc_c, wpack, biaspack, act, ctr, out);
}
